// AnswerDecoder_76330158784906
// MI455X (gfx1250) — matrix operand order repaired and checked by reading, not compiled, not run on hardware
//
#include <hip/hip_runtime.h>
#include <hip/hip_bf16.h>

typedef __bf16 bf16;
typedef __attribute__((ext_vector_type(16))) __bf16        v16bf;
typedef __attribute__((ext_vector_type(8)))  float         v8f;
typedef __attribute__((ext_vector_type(4)))  unsigned int  u32x4;

typedef __attribute__((address_space(1))) const u32x4* gc4p;  // global const 16B
typedef __attribute__((address_space(1))) float*       gfp;   // global f32
typedef __attribute__((address_space(1))) bf16*        gbp;   // global bf16

#define B_  64
#define T_  64
#define Qd  512
#define Ed  256
#define Hd  512
#define Vd  20000
#define START_TOK 1

__device__ __forceinline__ bf16 f2bf(float f) {
  unsigned u = __builtin_bit_cast(unsigned, f);
  unsigned r = (u + 0x7FFFu + ((u >> 16) & 1u)) >> 16;   // round-to-nearest-even
  unsigned short s = (unsigned short)r;
  return __builtin_bit_cast(bf16, s);
}

union Frag { v16bf v; u32x4 u[2]; };

// A fragment: lanes 0-15 row=m0+lane hold K {0..7,16..23}; lanes 16-31 same rows,
// K {8..15,24..31}.  Per-lane base p = row*lda + half*8; chunks at byte off 0 and 32.
__device__ __forceinline__ const bf16* a_base(const bf16* __restrict__ A, int lda,
                                              int m0, int lane) {
  return A + (size_t)(m0 + (lane & 15)) * lda + (lane >> 4) * 8;
}
__device__ __forceinline__ v16bf load_a_frag(const bf16* p) {
  Frag f;
  f.u[0] = ((gc4p)p)[0];   // K pairs      (bytes 0..15)
  f.u[1] = ((gc4p)p)[2];   // K pairs +16  (bytes 32..47)
  return f.v;
}
// B fragment (W row-major NxK, i.e. W[n][k]): lanes 0-15 col=n0+lane hold K {0..7,16..23};
// lanes 16-31 hold K {8..15,24..31}.  Chunks at byte off 0 and 32.
__device__ __forceinline__ const bf16* b_base(const bf16* __restrict__ W, int ldb,
                                              int n0, int lane) {
  return W + (size_t)(n0 + (lane & 15)) * ldb + (lane >> 4) * 8;
}
__device__ __forceinline__ v16bf load_b_frag(const bf16* p) {
  Frag f;
  f.u[0] = ((gc4p)p)[0];   // bytes 0..15
  f.u[1] = ((gc4p)p)[2];   // bytes 32..47
  return f.v;
}

__device__ __forceinline__ v8f wmma_bf16(v16bf a, v16bf b, v8f c) {
  return __builtin_amdgcn_wmma_f32_16x16x32_bf16(false, a, false, b,
                                                 (short)0, c, false, false);
}

// Software-pipelined K loop, 32x64 wave tile: 2 A frags x 4 B frags -> 8 WMMAs
// per K step; next tile's 6 fragments preload before current WMMAs issue.
// K >= 64, K % 32 == 0.
__device__ __forceinline__ void kloop2(const bf16* ap0, const bf16* ap1,
                                       const bf16* bp0, const bf16* bp1,
                                       const bf16* bp2, const bf16* bp3, int K,
                                       v8f (&acc)[8])
{
  v16bf a0 = load_a_frag(ap0);
  v16bf a1 = load_a_frag(ap1);
  v16bf b0 = load_b_frag(bp0);
  v16bf b1 = load_b_frag(bp1);
  v16bf b2 = load_b_frag(bp2);
  v16bf b3 = load_b_frag(bp3);
  for (int k0 = 32; k0 < K; k0 += 32) {
    ap0 += 32; ap1 += 32; bp0 += 32; bp1 += 32; bp2 += 32; bp3 += 32;
    v16bf na0 = load_a_frag(ap0);
    v16bf na1 = load_a_frag(ap1);
    v16bf nb0 = load_b_frag(bp0);
    v16bf nb1 = load_b_frag(bp1);
    v16bf nb2 = load_b_frag(bp2);
    v16bf nb3 = load_b_frag(bp3);
    acc[0] = wmma_bf16(a0, b0, acc[0]);
    acc[1] = wmma_bf16(a0, b1, acc[1]);
    acc[2] = wmma_bf16(a0, b2, acc[2]);
    acc[3] = wmma_bf16(a0, b3, acc[3]);
    acc[4] = wmma_bf16(a1, b0, acc[4]);
    acc[5] = wmma_bf16(a1, b1, acc[5]);
    acc[6] = wmma_bf16(a1, b2, acc[6]);
    acc[7] = wmma_bf16(a1, b3, acc[7]);
    a0 = na0; a1 = na1; b0 = nb0; b1 = nb1; b2 = nb2; b3 = nb3;
  }
  acc[0] = wmma_bf16(a0, b0, acc[0]);
  acc[1] = wmma_bf16(a0, b1, acc[1]);
  acc[2] = wmma_bf16(a0, b2, acc[2]);
  acc[3] = wmma_bf16(a0, b3, acc[3]);
  acc[4] = wmma_bf16(a1, b0, acc[4]);
  acc[5] = wmma_bf16(a1, b1, acc[5]);
  acc[6] = wmma_bf16(a1, b2, acc[6]);
  acc[7] = wmma_bf16(a1, b3, acc[7]);
}

// C/D layout: VGPR r: lanes 0-15 -> M=r, N=lane ; lanes 16-31 -> M=8+r, N=lane-16.
__device__ __forceinline__ void store_c(float* __restrict__ C, bf16* __restrict__ Cbf,
                                        int ldc, int m0, int n0, int lane, v8f acc,
                                        const float* __restrict__ bias) {
  int half = lane >> 4;
  int n = n0 + (lane & 15);
  float bv = bias ? bias[n] : 0.0f;
#pragma unroll
  for (int r = 0; r < 8; ++r) {
    size_t m = (size_t)(m0 + half * 8 + r);
    float val = acc[r] + bv;
    if (C)   *(gfp)(C + m * ldc + n)   = val;
    if (Cbf) *(gbp)(Cbf + m * ldc + n) = f2bf(val);
  }
}

// C[MxN] = A[MxK] * W[NxK]^T (+bias). M multiple of 64 via grid.y, N % 16 == 0.
// 256 threads = 8 waves as 2(M) x 4(N); WG tile 64x256, wave tile 32x64.
// Ragged-N subtiles clamped to column N-16 (redundant compute, never stored).
__global__ __launch_bounds__(256) void wmma_gemm(
    const bf16* __restrict__ A, int lda,
    const bf16* __restrict__ W, int ldb,
    float* __restrict__ C, bf16* __restrict__ Cbf, int ldc,
    int N, int K, const float* __restrict__ bias)
{
  int lane = threadIdx.x & 31;
  int wave = threadIdx.x >> 5;
  int m0 = blockIdx.y * 64 + (wave & 1) * 32;
  int nbase = blockIdx.x * 256 + (wave >> 1) * 64;

  int nc0 = (nbase +  0 < N) ? nbase +  0 : (N - 16);
  int nc1 = (nbase + 16 < N) ? nbase + 16 : (N - 16);
  int nc2 = (nbase + 32 < N) ? nbase + 32 : (N - 16);
  int nc3 = (nbase + 48 < N) ? nbase + 48 : (N - 16);

  v8f acc[8] = {};
  kloop2(a_base(A, lda, m0, lane), a_base(A, lda, m0 + 16, lane),
         b_base(W, ldb, nc0, lane), b_base(W, ldb, nc1, lane),
         b_base(W, ldb, nc2, lane), b_base(W, ldb, nc3, lane),
         K, acc);

#pragma unroll
  for (int mi = 0; mi < 2; ++mi) {
#pragma unroll
    for (int j = 0; j < 4; ++j) {
      int n0 = nbase + j * 16;
      if (__builtin_amdgcn_readfirstlane(n0 < N ? 1 : 0))   // scalar branch
        store_c(C, Cbf, ldc, m0 + mi * 16, n0, lane, acc[mi * 4 + j], bias);
    }
  }
}

// gates[64 x 2048] = X[64xE]*Wih^T + Hb[64xH]*Whh^T + bih + bhh  (N=2048 exact)
__global__ __launch_bounds__(256) void gates_gemm(
    const bf16* __restrict__ X, const bf16* __restrict__ Hb,
    const bf16* __restrict__ Wih, const bf16* __restrict__ Whh,
    const float* __restrict__ bih, const float* __restrict__ bhh,
    float* __restrict__ G)
{
  int lane = threadIdx.x & 31;
  int wave = threadIdx.x >> 5;
  int m0 = (wave & 1) * 32;
  int nbase = blockIdx.x * 256 + (wave >> 1) * 64;

  v8f acc[8] = {};
  kloop2(a_base(X, Ed, m0, lane), a_base(X, Ed, m0 + 16, lane),
         b_base(Wih, Ed, nbase +  0, lane), b_base(Wih, Ed, nbase + 16, lane),
         b_base(Wih, Ed, nbase + 32, lane), b_base(Wih, Ed, nbase + 48, lane),
         Ed, acc);
  kloop2(a_base(Hb, Hd, m0, lane), a_base(Hb, Hd, m0 + 16, lane),
         b_base(Whh, Hd, nbase +  0, lane), b_base(Whh, Hd, nbase + 16, lane),
         b_base(Whh, Hd, nbase + 32, lane), b_base(Whh, Hd, nbase + 48, lane),
         Hd, acc);

  int half = lane >> 4;
#pragma unroll
  for (int mi = 0; mi < 2; ++mi) {
#pragma unroll
    for (int j = 0; j < 4; ++j) {
      int n = nbase + j * 16 + (lane & 15);
      float bv = bih[n] + bhh[n];
#pragma unroll
      for (int r = 0; r < 8; ++r) {
        int m = m0 + mi * 16 + half * 8 + r;
        *(gfp)(G + (size_t)m * (4 * Hd) + n) = acc[mi * 4 + j][r] + bv;
      }
    }
  }
}

__global__ void lstm_pointwise(const float* __restrict__ G, float* __restrict__ c,
                               bf16* __restrict__ hbf, bf16* __restrict__ Hall, int t)
{
  int i = blockIdx.x * blockDim.x + threadIdx.x;
  if (i >= B_ * Hd) return;
  int b = i / Hd, j = i - b * Hd;
  const float* g = G + (size_t)b * 4 * Hd;
  float ig = 1.0f / (1.0f + __expf(-g[j]));
  float fg = 1.0f / (1.0f + __expf(-g[Hd + j]));
  float gg = tanhf(g[2 * Hd + j]);
  float og = 1.0f / (1.0f + __expf(-g[3 * Hd + j]));
  float cn = fg * c[i] + ig * gg;
  float hn = og * tanhf(cn);
  c[i] = cn;
  bf16 hb = f2bf(hn);
  hbf[i] = hb;
  Hall[((size_t)b * T_ + t) * Hd + j] = hb;   // row b*T+t -> (B,T,V) output order
}

__global__ void gather_xs(const float* __restrict__ emb, const int* __restrict__ answers,
                          bf16* __restrict__ xs)
{
  int i = blockIdx.x * blockDim.x + threadIdx.x;   // over T*B*E
  if (i >= T_ * B_ * Ed) return;
  int e = i % Ed;
  int tb = i / Ed;
  int b = tb % B_;
  int t = tb / B_;
  int tok = (t == 0) ? START_TOK : answers[b * T_ + (t - 1)];
  xs[i] = f2bf(emb[(size_t)tok * Ed + e]);
}

__global__ void cvt_f32_bf16(const float* __restrict__ src, bf16* __restrict__ dst, int n)
{
  int i = blockIdx.x * blockDim.x + threadIdx.x;
  if (i < n) dst[i] = f2bf(src[i]);
}

extern "C" void kernel_launch(void* const* d_in, const int* in_sizes, int n_in,
                              void* d_out, int out_size, void* d_ws, size_t ws_size,
                              hipStream_t stream)
{
  const float* q       = (const float*)d_in[0];
  const float* emb     = (const float*)d_in[1];
  const float* W_h     = (const float*)d_in[2];
  const float* W_c     = (const float*)d_in[3];
  const float* W_ih    = (const float*)d_in[4];
  const float* W_hh    = (const float*)d_in[5];
  const float* b_ih    = (const float*)d_in[6];
  const float* b_hh    = (const float*)d_in[7];
  const float* W_vocab = (const float*)d_in[8];
  const float* b_vocab = (const float*)d_in[9];
  const int*   answers = (const int*)d_in[10];
  float* out = (float*)d_out;

  char* ws = (char*)d_ws;
  size_t off = 0;
  auto alloc = [&](size_t bytes) -> void* {
    void* p = ws + off;
    off = (off + bytes + 255) & ~(size_t)255;
    return p;
  };
  bf16*  Wih_bf = (bf16*)alloc((size_t)4 * Hd * Ed * 2);
  bf16*  Whh_bf = (bf16*)alloc((size_t)4 * Hd * Hd * 2);
  bf16*  Wh_bf  = (bf16*)alloc((size_t)Hd * Qd * 2);
  bf16*  Wc_bf  = (bf16*)alloc((size_t)Hd * Qd * 2);
  bf16*  Wv_bf  = (bf16*)alloc((size_t)Vd * Hd * 2);
  bf16*  q_bf   = (bf16*)alloc((size_t)B_ * Qd * 2);
  bf16*  xs_bf  = (bf16*)alloc((size_t)T_ * B_ * Ed * 2);
  bf16*  h_bf   = (bf16*)alloc((size_t)B_ * Hd * 2);
  float* c_f32  = (float*)alloc((size_t)B_ * Hd * 4);
  float* gates  = (float*)alloc((size_t)B_ * 4 * Hd * 4);
  bf16*  Hall   = (bf16*)alloc((size_t)B_ * T_ * Hd * 2);

  auto cvt = [&](const float* s, bf16* d, int n) {
    cvt_f32_bf16<<<(n + 255) / 256, 256, 0, stream>>>(s, d, n);
  };
  cvt(W_ih,    Wih_bf, 4 * Hd * Ed);
  cvt(W_hh,    Whh_bf, 4 * Hd * Hd);
  cvt(W_h,     Wh_bf,  Hd * Qd);
  cvt(W_c,     Wc_bf,  Hd * Qd);
  cvt(W_vocab, Wv_bf,  Vd * Hd);
  cvt(q,       q_bf,   B_ * Qd);
  gather_xs<<<(T_ * B_ * Ed + 255) / 256, 256, 0, stream>>>(emb, answers, xs_bf);

  // h0 = q @ W_h^T  (bf16 copy only), c0 = q @ W_c^T (fp32); M=64,N=512,K=512
  wmma_gemm<<<dim3(2, 1), 256, 0, stream>>>(q_bf, Qd, Wh_bf, Qd,
                                            nullptr, h_bf, Hd, Hd, Qd, nullptr);
  wmma_gemm<<<dim3(2, 1), 256, 0, stream>>>(q_bf, Qd, Wc_bf, Qd,
                                            c_f32, nullptr, Hd, Hd, Qd, nullptr);

  for (int t = 0; t < T_; ++t) {
    gates_gemm<<<8, 256, 0, stream>>>(xs_bf + (size_t)t * B_ * Ed, h_bf,
                                      Wih_bf, Whh_bf, b_ih, b_hh, gates);
    lstm_pointwise<<<(B_ * Hd + 255) / 256, 256, 0, stream>>>(gates, c_f32, h_bf, Hall, t);
  }

  // Batched logits: [4096 x 20000] = Hall[4096x512] @ W_vocab^T + b_vocab
  wmma_gemm<<<dim3((Vd + 255) / 256, (B_ * T_) / 64), 256, 0, stream>>>(
      Hall, Hd, Wv_bf, Hd, out, nullptr, Vd, Vd, Hd, b_vocab);
}
